// MultiHeadRelativeAttention_4930622456076
// MI455X (gfx1250) — matrix operand order repaired and checked by reading, not compiled, not run on hardware
//
#include <hip/hip_runtime.h>

// ---------------------------------------------------------------------------
// MultiHeadRelativeAttention (Transformer-XL style) for MI455X / gfx1250.
// S=256, E=1024, H=16, HD=64.
// bf16 WMMA (v_wmma_f32_16x16x32_bf16) with 16x64 tiles/wave for all GEMMs;
// non-temporal fp32 streaming for the 256MB s_k term (memory-bound ~11us at
// 23.3 TB/s; everything else hides under it / lives in L2).
// ---------------------------------------------------------------------------

#define S_LEN 256
#define EMB   1024
#define NHEAD 16
#define HDIM  64

typedef __attribute__((ext_vector_type(16))) __bf16 v16bf;
typedef __attribute__((ext_vector_type(8)))  __bf16 v8bf;
typedef __attribute__((ext_vector_type(8)))  float  v8f;
typedef __attribute__((ext_vector_type(4)))  float  v4f;

__device__ __forceinline__ unsigned short f2bf(float f) {
    unsigned u = __float_as_uint(f);
    u = (u + 0x7FFFu + ((u >> 16) & 1u)) >> 16;   // round-to-nearest-even
    return (unsigned short)u;
}

// ---------------------------------------------------------------------------
// fp32 -> bf16 elementwise (n multiple of 4)
// ---------------------------------------------------------------------------
__global__ void f32_to_bf16_kernel(const float* __restrict__ in,
                                   unsigned short* __restrict__ out, int n) {
    int idx = (blockIdx.x * blockDim.x + threadIdx.x) * 4;
    if (idx >= n) return;
    v4f v = *(const v4f*)(in + idx);
    out[idx + 0] = f2bf(v[0]);
    out[idx + 1] = f2bf(v[1]);
    out[idx + 2] = f2bf(v[2]);
    out[idx + 3] = f2bf(v[3]);
}

// ---------------------------------------------------------------------------
// bf16 transpose: in (S_LEN, EMB) -> outT (EMB, S_LEN)
// ---------------------------------------------------------------------------
__global__ void transpose_bf16_kernel(const unsigned short* __restrict__ in,
                                      unsigned short* __restrict__ outT) {
    int idx = blockIdx.x * blockDim.x + threadIdx.x;   // 0 .. S*E-1
    int r = idx >> 10;                                 // row (0..255)
    int c = idx & 1023;                                // col (0..1023)
    outT[(size_t)c * S_LEN + r] = in[idx];
}

// ---------------------------------------------------------------------------
// WMMA fragment helpers (wave32, 16x16 WMMA, 16x64 tile per wave)
// A (MxK row-major) 16-bit layout: lane l: M=l&15, group g=l>>4;
//   VGPR0..7 hold K = g*16+0..15 (contiguous 16; this code's order, same for A and B, not the ISA table order).
// B for NT (B is NxK row-major): lane l: N=l&15, g=l>>4;
//   VGPR0..7 hold K = g*16+0..15 (contiguous 16; this code's order, same for A and B, not the ISA table order).
// C 16x16 f32: lane l: N=l&15, g=l>>4; VGPR r holds M = g*8 + r.
// ---------------------------------------------------------------------------
__device__ __forceinline__ v16bf load_a_frag(const __bf16* __restrict__ arow, int k0) {
    v8bf c0 = *(const v8bf*)(arow + k0);
    v8bf c1 = *(const v8bf*)(arow + k0 + 8);
    v16bf a;
#pragma unroll
    for (int t = 0; t < 8; ++t) { a[t] = c0[t]; a[8 + t] = c1[t]; }
    return a;
}

__device__ __forceinline__ void store_c(v8f acc, float* Cf, unsigned short* Cb,
                                        size_t cbase, int ldc, int tm, int tn,
                                        const float* bias, int lane) {
    int n = lane & 15, g = lane >> 4;
    float b = bias ? bias[tn + n] : 0.0f;
#pragma unroll
    for (int r = 0; r < 8; ++r) {
        int m = g * 8 + r;
        float val = acc[r] + b;
        size_t idx = cbase + (size_t)(tm + m) * ldc + tn + n;
        if (Cf) Cf[idx] = val;
        if (Cb) Cb[idx] = f2bf(val);
    }
}

// ---------------------------------------------------------------------------
// NT GEMM, 16x64 tile per wave: C[m,n] = sum_k A[m,k] * B[n,k]
// A fragment reused across 4 N-subtiles -> 4 WMMAs per K-step.
// Optional per-head batching via blockIdx.y with element-offset strides.
// N must be a multiple of 64; grids sized so every wave is full (EXEC all-1).
// ---------------------------------------------------------------------------
__global__ void gemm_nt64_bf16(const __bf16* __restrict__ A, int lda, int aHS,
                               const __bf16* __restrict__ B, int ldb, int bHS,
                               float* __restrict__ Cf, unsigned short* __restrict__ Cb,
                               int ldc, int cHS, int M, int N, int K,
                               const float* __restrict__ bias) {
    int lane   = threadIdx.x & 31;
    int wave   = (blockIdx.x * blockDim.x + threadIdx.x) >> 5;
    int tilesN = N >> 6;                       // 64-wide tiles
    int tiles  = (M >> 4) * tilesN;
    if (wave >= tiles) return;                 // wave-uniform exit
    int tm = (wave / tilesN) << 4;
    int tn = (wave % tilesN) << 6;
    int h  = blockIdx.y;
    A += (size_t)h * aHS;
    B += (size_t)h * bHS;

    int mr = lane & 15, g = lane >> 4;
    const __bf16* arow = A + (size_t)(tm + mr) * lda + g * 16;
    const __bf16* brow = B + (size_t)(tn + mr) * ldb + g * 16;
    size_t bstep = (size_t)16 * ldb;

    v8f acc0 = {}, acc1 = {}, acc2 = {}, acc3 = {};
    for (int k0 = 0; k0 < K; k0 += 32) {
        v16bf af = load_a_frag(arow, k0);
        v16bf b0 = *(const v16bf*)(brow + k0);
        v16bf b1 = *(const v16bf*)(brow + bstep + k0);
        v16bf b2 = *(const v16bf*)(brow + 2 * bstep + k0);
        v16bf b3 = *(const v16bf*)(brow + 3 * bstep + k0);
        acc0 = __builtin_amdgcn_wmma_f32_16x16x32_bf16(false, af, false, b0, (short)0, acc0, false, false);
        acc1 = __builtin_amdgcn_wmma_f32_16x16x32_bf16(false, af, false, b1, (short)0, acc1, false, false);
        acc2 = __builtin_amdgcn_wmma_f32_16x16x32_bf16(false, af, false, b2, (short)0, acc2, false, false);
        acc3 = __builtin_amdgcn_wmma_f32_16x16x32_bf16(false, af, false, b3, (short)0, acc3, false, false);
    }
    size_t cbase = (size_t)h * cHS;
    store_c(acc0, Cf, Cb, cbase, ldc, tm, tn,      bias, lane);
    store_c(acc1, Cf, Cb, cbase, ldc, tm, tn + 16, bias, lane);
    store_c(acc2, Cf, Cb, cbase, ldc, tm, tn + 32, bias, lane);
    store_c(acc3, Cf, Cb, cbase, ldc, tm, tn + 48, bias, lane);
}

// ---------------------------------------------------------------------------
// Fused score kernel: one wave per (h,i) row.
//  a2[j]   = dot(new_q[h,i,:], s_k[i*S+j, h*64:...])   -- 256MB fp32 NT stream
//  a3[i,j] = (j<=i) ? a3raw[h,i, 255-i+j] : 0          -- analytic rel_shift+tril
//  w = (a1+a2+a3)*HD^-0.5 ; masked -> 1e-30 ; wave softmax ; bf16 scores out
// ---------------------------------------------------------------------------
__global__ void score_kernel(const float* __restrict__ NQf,
                             const float* __restrict__ a1,
                             const float* __restrict__ a3raw,
                             const int*   __restrict__ mask,
                             const float* __restrict__ s_k,
                             unsigned short* __restrict__ scores) {
    __shared__ float nq[8][HDIM];
    int lane = threadIdx.x & 31;
    int w    = threadIdx.x >> 5;
    int row  = blockIdx.x * 8 + w;           // 0..H*S-1
    int h    = row >> 8;
    int i    = row & 255;

    // stage new_q row (64 floats) into LDS
    nq[w][lane * 2]     = NQf[(size_t)i * EMB + h * HDIM + lane * 2];
    nq[w][lane * 2 + 1] = NQf[(size_t)i * EMB + h * HDIM + lane * 2 + 1];
    __syncthreads();

    const float scale = 0.125f;              // HD^-0.5
    float wv[8];
    float mmax = -3.4e38f;

#pragma unroll
    for (int it = 0; it < 8; ++it) {
        int j = it * 32 + lane;
        const float* skp = s_k + ((size_t)i * S_LEN + j) * EMB + h * HDIM;
        float a2 = 0.0f;
#pragma unroll
        for (int t = 0; t < 16; ++t) {
            // read-once 256MB stream: non-temporal so it doesn't evict L2
            v4f vv = __builtin_nontemporal_load((const v4f*)(skp + 4 * t));
            a2 += vv[0] * nq[w][4 * t]     + vv[1] * nq[w][4 * t + 1]
                + vv[2] * nq[w][4 * t + 2] + vv[3] * nq[w][4 * t + 3];
        }
        size_t base = (size_t)h * (S_LEN * S_LEN) + (size_t)i * S_LEN + j;
        float v1 = a1[base];
        float v3 = (j <= i)
                 ? a3raw[(size_t)h * (S_LEN * S_LEN) + (size_t)i * S_LEN + (255 - i + j)]
                 : 0.0f;
        float val = (v1 + a2 + v3) * scale;
        if (mask[base]) val = 1e-30f;
        wv[it] = val;
        mmax = fmaxf(mmax, val);
    }
#pragma unroll
    for (int o = 16; o >= 1; o >>= 1) mmax = fmaxf(mmax, __shfl_xor(mmax, o, 32));

    float sum = 0.0f;
#pragma unroll
    for (int it = 0; it < 8; ++it) { wv[it] = __expf(wv[it] - mmax); sum += wv[it]; }
#pragma unroll
    for (int o = 16; o >= 1; o >>= 1) sum += __shfl_xor(sum, o, 32);
    float inv = 1.0f / sum;

#pragma unroll
    for (int it = 0; it < 8; ++it) {
        int j = it * 32 + lane;
        scores[(size_t)h * (S_LEN * S_LEN) + (size_t)i * S_LEN + j] = f2bf(wv[it] * inv);
    }
}

// ---------------------------------------------------------------------------
// Host launcher
// ---------------------------------------------------------------------------
extern "C" void kernel_launch(void* const* d_in, const int* in_sizes, int n_in,
                              void* d_out, int out_size, void* d_ws, size_t ws_size,
                              hipStream_t stream) {
    (void)in_sizes; (void)n_in; (void)out_size; (void)ws_size;

    const float* q    = (const float*)d_in[0];
    const float* k    = (const float*)d_in[1];
    const float* v    = (const float*)d_in[2];
    const int*   mask = (const int*)  d_in[3];
    const float* p    = (const float*)d_in[4];
    const float* s_q  = (const float*)d_in[5];
    const float* s_k  = (const float*)d_in[6];
    const float* Wq   = (const float*)d_in[7];
    const float* Wk   = (const float*)d_in[8];
    const float* Wv   = (const float*)d_in[9];
    const float* Wr   = (const float*)d_in[10];
    const float* Wo   = (const float*)d_in[11];
    float* out = (float*)d_out;

    const int SE  = S_LEN * EMB;        // 262144
    const int EE  = EMB * EMB;          // 1048576
    const int HSS = NHEAD * S_LEN * S_LEN;

    // ---- workspace carve-out (256B aligned) ----
    char*  ws  = (char*)d_ws;
    size_t off = 0;
    auto take = [&](size_t bytes) -> char* {
        char* ptr = ws + off;
        off += (bytes + 255) & ~(size_t)255;
        return ptr;
    };
    unsigned short* qbf  = (unsigned short*)take((size_t)SE * 2);
    unsigned short* kbf  = (unsigned short*)take((size_t)SE * 2);
    unsigned short* vbf  = (unsigned short*)take((size_t)SE * 2);
    unsigned short* pbf  = (unsigned short*)take((size_t)SE * 2);
    unsigned short* Wqb  = (unsigned short*)take((size_t)EE * 2);
    unsigned short* Wkb  = (unsigned short*)take((size_t)EE * 2);
    unsigned short* Wvb  = (unsigned short*)take((size_t)EE * 2);
    unsigned short* Wrb  = (unsigned short*)take((size_t)EE * 2);
    unsigned short* Wob  = (unsigned short*)take((size_t)EE * 2);
    float*          NQf  = (float*)         take((size_t)SE * 4);
    unsigned short* NQb  = (unsigned short*)take((size_t)SE * 2);
    unsigned short* Kb   = (unsigned short*)take((size_t)SE * 2);
    unsigned short* Vb   = (unsigned short*)take((size_t)SE * 2);
    unsigned short* Rb   = (unsigned short*)take((size_t)SE * 2);
    unsigned short* Vtb  = (unsigned short*)take((size_t)SE * 2);   // V^T (E,S)
    float*          a1   = (float*)         take((size_t)HSS * 4);
    float*          a3r  = (float*)         take((size_t)HSS * 4);
    unsigned short* scb  = (unsigned short*)take((size_t)HSS * 2);
    unsigned short* AOb  = (unsigned short*)take((size_t)SE * 2);

    // ---- 1) fp32 -> bf16 conversions ----
    {
        dim3 blk(256);
        f32_to_bf16_kernel<<<dim3(SE / 1024), blk, 0, stream>>>(q,  qbf, SE);
        f32_to_bf16_kernel<<<dim3(SE / 1024), blk, 0, stream>>>(k,  kbf, SE);
        f32_to_bf16_kernel<<<dim3(SE / 1024), blk, 0, stream>>>(v,  vbf, SE);
        f32_to_bf16_kernel<<<dim3(SE / 1024), blk, 0, stream>>>(p,  pbf, SE);
        f32_to_bf16_kernel<<<dim3(EE / 1024), blk, 0, stream>>>(Wq, Wqb, EE);
        f32_to_bf16_kernel<<<dim3(EE / 1024), blk, 0, stream>>>(Wk, Wkb, EE);
        f32_to_bf16_kernel<<<dim3(EE / 1024), blk, 0, stream>>>(Wv, Wvb, EE);
        f32_to_bf16_kernel<<<dim3(EE / 1024), blk, 0, stream>>>(Wr, Wrb, EE);
        f32_to_bf16_kernel<<<dim3(EE / 1024), blk, 0, stream>>>(Wo, Wob, EE);
    }

    // ---- 2) projections: (256x1024)@(1024x1024)^T, 16 M-tiles x 16 N64-tiles
    //         = 256 waves -> 32 blocks ----
    {
        dim3 blk(256), grd(32, 1);
        // new_q = q@Wq^T + s_q (bias); keep fp32 (for a2) and bf16 (for WMMA)
        gemm_nt64_bf16<<<grd, blk, 0, stream>>>((const __bf16*)qbf, EMB, 0,
                                                (const __bf16*)Wqb, EMB, 0,
                                                NQf, NQb, EMB, 0,
                                                S_LEN, EMB, EMB, s_q);
        gemm_nt64_bf16<<<grd, blk, 0, stream>>>((const __bf16*)kbf, EMB, 0,
                                                (const __bf16*)Wkb, EMB, 0,
                                                nullptr, Kb, EMB, 0,
                                                S_LEN, EMB, EMB, nullptr);
        gemm_nt64_bf16<<<grd, blk, 0, stream>>>((const __bf16*)vbf, EMB, 0,
                                                (const __bf16*)Wvb, EMB, 0,
                                                nullptr, Vb, EMB, 0,
                                                S_LEN, EMB, EMB, nullptr);
        gemm_nt64_bf16<<<grd, blk, 0, stream>>>((const __bf16*)pbf, EMB, 0,
                                                (const __bf16*)Wrb, EMB, 0,
                                                nullptr, Rb, EMB, 0,
                                                S_LEN, EMB, EMB, nullptr);
    }

    // ---- 2b) V^T for the attn_out GEMM ----
    transpose_bf16_kernel<<<dim3(SE / 256), dim3(256), 0, stream>>>(Vb, Vtb);

    // ---- 3) per-head a1 = NQ@K^T, a3raw = NQ@R^T (K=64):
    //         16 M-tiles x 4 N64-tiles = 64 waves -> 8 blocks, grid.y = heads ----
    {
        dim3 blk(256), grd(8, NHEAD);
        gemm_nt64_bf16<<<grd, blk, 0, stream>>>((const __bf16*)NQb, EMB, HDIM,
                                                (const __bf16*)Kb,  EMB, HDIM,
                                                a1, nullptr, S_LEN, S_LEN * S_LEN,
                                                S_LEN, S_LEN, HDIM, nullptr);
        gemm_nt64_bf16<<<grd, blk, 0, stream>>>((const __bf16*)NQb, EMB, HDIM,
                                                (const __bf16*)Rb,  EMB, HDIM,
                                                a3r, nullptr, S_LEN, S_LEN * S_LEN,
                                                S_LEN, S_LEN, HDIM, nullptr);
    }

    // ---- 4) fused a2 + rel_shift + mask + softmax: 4096 rows, 8 waves/block ----
    score_kernel<<<dim3(512), dim3(256), 0, stream>>>(NQf, a1, a3r, mask, s_k, scb);

    // ---- 5) attn_out = score @ V  (as NT with V^T): per head
    //         16 M-tiles x 1 N64-tile = 16 waves -> 2 blocks, grid.y = heads ----
    {
        dim3 blk(256), grd(2, NHEAD);
        gemm_nt64_bf16<<<grd, blk, 0, stream>>>((const __bf16*)scb, S_LEN, S_LEN * S_LEN,
                                                (const __bf16*)Vtb, S_LEN, HDIM * S_LEN,
                                                nullptr, AOb, EMB, HDIM,
                                                S_LEN, HDIM, S_LEN, nullptr);
    }

    // ---- 6) out = attn_out @ Wo^T ----
    {
        dim3 blk(256), grd(32, 1);
        gemm_nt64_bf16<<<grd, blk, 0, stream>>>((const __bf16*)AOb, EMB, 0,
                                                (const __bf16*)Wob, EMB, 0,
                                                out, nullptr, EMB, 0,
                                                S_LEN, EMB, EMB, nullptr);
    }
}
